// VideoDecoder_41798621724871
// MI455X (gfx1250) — matrix operand order repaired and checked by reading, not compiled, not run on hardware
//
#include <hip/hip_runtime.h>

// ---------------------------------------------------------------------------
// CDNA5 (gfx1250) wave32 WMMA implementation of the LSTM->CNN video decoder.
// All GEMM-like math runs through v_wmma_f32_16x16x32_bf16 (bf16 in, f32 acc).
// Deconvs (k=4,s=2,p=1) are implicit GEMMs per output parity class.
// ---------------------------------------------------------------------------

typedef __bf16 bf16_t;
typedef __attribute__((ext_vector_type(16))) __bf16 v16bf;
typedef __attribute__((ext_vector_type(8)))  __bf16 v8bf;
typedef __attribute__((ext_vector_type(8)))  float  v8f;

__device__ __forceinline__ bf16_t f2bf(float f) { return (bf16_t)f; }
__device__ __forceinline__ float  bf2f(bf16_t b) { return (float)b; }
__device__ __forceinline__ float  sigmoidf_(float x) { return 1.0f / (1.0f + __expf(-x)); }
__device__ __forceinline__ float  leaky_(float x) { return x > 0.0f ? x : 0.2f * x; }

__device__ __forceinline__ v16bf cat8(v8bf lo, v8bf hi) {
  return __builtin_shufflevector(lo, hi, 0,1,2,3,4,5,6,7,8,9,10,11,12,13,14,15);
}
__device__ __forceinline__ v16bf zero16() {
  v16bf r;
#pragma unroll
  for (int i = 0; i < 16; ++i) r[i] = (bf16_t)0.0f;
  return r;
}
__device__ __forceinline__ v8f wmma_bf16(v16bf a, v16bf b, v8f c) {
  // 8-arg form: (neg_a, A, neg_b, B, c_mod, C, reuse_a, reuse_b)
  return __builtin_amdgcn_wmma_f32_16x16x32_bf16(false, a, false, b, (short)0, c, false, false);
}

// A-fragment (16x32 bf16): per-lane one contiguous 16-element run at
// k0 + (lane>>4)*16 (contiguous order, matching B).
__device__ __forceinline__ v16bf load_a(const bf16_t* rowp, int ks) {
  return cat8(*(const v8bf*)(rowp + ks), *(const v8bf*)(rowp + ks + 8));
}

// ---------------------------------------------------------------------------
// Weight packing kernels (run once per launch; L2 resident afterwards)
// ---------------------------------------------------------------------------
__global__ void k_cvt(const float* __restrict__ src, bf16_t* __restrict__ dst, int n) {
  int i = blockIdx.x * 256 + threadIdx.x;
  if (i < n) dst[i] = f2bf(src[i]);
}

__global__ void k_pack_wg(const float* __restrict__ w_ih, const float* __restrict__ w_hh,
                          bf16_t* __restrict__ wg) {
  int i = blockIdx.x * 256 + threadIdx.x;          // over 1024*512
  if (i >= 1024 * 512) return;
  int n = i >> 9, k = i & 511;
  float v = (k < 256) ? w_ih[n * 256 + k] : w_hh[n * 256 + (k - 256)];
  wg[i] = f2bf(v);
}

// ConvT(k=4,s=2,p=1): parity p=(py,px); tap t=(ty,tx):
//   py==0: ky = ty?3:1, diy = ty?-1:0 ;  py==1: ky = ty?2:0, diy = ty?0:1
__global__ void k_pack_deconv(const float* __restrict__ dw, bf16_t* __restrict__ bp,
                              int Cin, int Cout) {
  int K4 = Cin * 4;
  int total = 4 * Cout * K4;
  int i = blockIdx.x * 256 + threadIdx.x;
  if (i >= total) return;
  int ci = i % Cin;
  int t  = (i / Cin) & 3;
  int co = (i / K4) % Cout;
  int p  = i / (K4 * Cout);
  int py = p >> 1, px = p & 1, ty = t >> 1, tx = t & 1;
  int ky = py ? (ty ? 2 : 0) : (ty ? 3 : 1);
  int kx = px ? (tx ? 2 : 0) : (tx ? 3 : 1);
  bp[i] = f2bf(dw[((ci * Cout + co) * 4 + ky) * 4 + kx]);
}

__global__ void k_pack_final(const float* __restrict__ fw, float* __restrict__ fwp) {
  int i = blockIdx.x * 256 + threadIdx.x;          // 4 parities * 4 taps * 32 ci
  if (i >= 512) return;
  int ci = i & 31, t = (i >> 5) & 3, p = i >> 7;
  int py = p >> 1, px = p & 1, ty = t >> 1, tx = t & 1;
  int ky = py ? (ty ? 2 : 0) : (ty ? 3 : 1);
  int kx = px ? (tx ? 2 : 0) : (tx ? 3 : 1);
  fwp[i] = fw[ci * 16 + ky * 4 + kx];
}

__global__ void k_zero(float* __restrict__ p, int n) {
  int i = blockIdx.x * 256 + threadIdx.x;
  if (i < n) p[i] = 0.0f;
}

// ---------------------------------------------------------------------------
// Persistent single-workgroup LSTM decoder (1024 threads = 32 waves, one WGP).
// LDS: gates 32x1024 f32 (128K) | xh 32x512 bf16 (32K) | c 32x256 f32 (32K).
// ---------------------------------------------------------------------------
__global__ void __launch_bounds__(1024) k_lstm(
    const float* __restrict__ z, const float* __restrict__ init_inp,
    const bf16_t* __restrict__ whc, const float* __restrict__ fc_hidden_b,
    const float* __restrict__ fc_cell_b,
    const bf16_t* __restrict__ wg, const float* __restrict__ b_ih,
    const float* __restrict__ b_hh, bf16_t* __restrict__ outs)
{
  extern __shared__ char smem[];
  float*  gates = (float*)smem;                               // 32 x 1024
  bf16_t* xh    = (bf16_t*)(smem + 32 * 1024 * 4);            // 32 x [x(256)|h(256)]
  float*  cst   = (float*)(smem + 32 * 1024 * 4 + 32 * 512 * 2); // 32 x 256
  bf16_t* zbf   = (bf16_t*)smem;                              // temp, aliases gates

  const int tid = threadIdx.x;
  const int lane = tid & 31, wave = tid >> 5;
  const int khalf = lane >> 4, l15 = lane & 15;

  for (int i = tid; i < 32 * 256; i += 1024) {
    zbf[i] = f2bf(z[i]);
    xh[(i >> 8) * 512 + (i & 255)] = f2bf(init_inp[i & 255]);  // x0 = init_inp bcast
  }
  __syncthreads();

  // [h0|c0] = z @ [fc_hidden_w ; fc_cell_w]^T   (M=32, N=512, K=256)
#pragma unroll
  for (int it = 0; it < 2; ++it) {
    int tl = wave + it * 32;
    int m0 = (tl & 1) * 16, n0 = (tl >> 1) * 16;
    v8f acc = {};
    const bf16_t* arow = zbf + (m0 + l15) * 256 + khalf * 16;
    const bf16_t* brow = whc + (size_t)(n0 + l15) * 256 + khalf * 16;
    for (int ks = 0; ks < 256; ks += 32)
      acc = wmma_bf16(load_a(arow, ks), *(const v16bf*)(brow + ks), acc);
    int col = n0 + l15;
#pragma unroll
    for (int r = 0; r < 8; ++r) {
      int row = m0 + khalf * 8 + r;
      float v = acc[r];
      if (col < 256) xh[row * 512 + 256 + col] = f2bf(v + fc_hidden_b[col]);
      else           cst[row * 256 + (col - 256)] = v + fc_cell_b[col - 256];
    }
  }
  __syncthreads();

  for (int t = 0; t < 20; ++t) {
    // gates = [x|h] @ [w_ih|w_hh]^T   (M=32, N=1024, K=512); 128 tiles / 32 waves
#pragma unroll
    for (int it = 0; it < 4; ++it) {
      int tl = wave + it * 32;
      int m0 = (tl & 1) * 16, n0 = (tl >> 1) * 16;
      v8f acc = {};
      const bf16_t* arow = xh + (m0 + l15) * 512 + khalf * 16;
      const bf16_t* brow = wg + (size_t)(n0 + l15) * 512 + khalf * 16;
      for (int ks = 0; ks < 512; ks += 32)
        acc = wmma_bf16(load_a(arow, ks), *(const v16bf*)(brow + ks), acc);
      int col = n0 + l15;
#pragma unroll
      for (int r = 0; r < 8; ++r)
        gates[(m0 + khalf * 8 + r) * 1024 + col] = acc[r];
    }
    __syncthreads();

    // elementwise LSTM cell (PyTorch gate order i,f,g,o); x_{t+1} = h_{t+1}
    for (int e = tid; e < 32 * 256; e += 1024) {
      int b = e >> 8, n = e & 255;
      const float* grow = gates + b * 1024;
      float gi = grow[n]       + b_ih[n]       + b_hh[n];
      float gf = grow[256 + n] + b_ih[256 + n] + b_hh[256 + n];
      float gg = grow[512 + n] + b_ih[512 + n] + b_hh[512 + n];
      float go = grow[768 + n] + b_ih[768 + n] + b_hh[768 + n];
      float cn = sigmoidf_(gf) * cst[e] + sigmoidf_(gi) * tanhf(gg);
      float hn = sigmoidf_(go) * tanhf(cn);
      cst[e] = cn;
      bf16_t hb = f2bf(hn);
      xh[b * 512 + n] = hb;          // next x
      xh[b * 512 + 256 + n] = hb;    // next h
      outs[(size_t)(t * 32 + b) * 256 + n] = hb;   // row = t*32 + b
    }
    __syncthreads();
  }
}

// ---------------------------------------------------------------------------
// Generic WMMA GEMM: C = leaky(A[M x K] * Bw[N x K]^T + bias).
// Block 256 (8 waves), tile 64(M) x 128(N). mode 1: (t,b)->(b,t) row remap
// (fc_out/feats). mode 2: column remap NCHW(256,4,4)->NHWC store (cnn_fc).
// ---------------------------------------------------------------------------
__global__ void __launch_bounds__(256) k_gemm(
    const bf16_t* __restrict__ A, const bf16_t* __restrict__ Bw,
    const float* __restrict__ bias, bf16_t* __restrict__ out,
    int N, int K, int mode)
{
  const int lane = threadIdx.x & 31, wave = threadIdx.x >> 5;
  const int khalf = lane >> 4, l15 = lane & 15;
  const int m0 = blockIdx.x * 64 + (wave & 3) * 16;
  const int n0base = blockIdx.y * 128 + (wave >> 2) * 64;

  v8f zf = {};
  v8f acc[4];
#pragma unroll
  for (int i = 0; i < 4; ++i) acc[i] = zf;

  const bf16_t* arow = A + (size_t)(m0 + l15) * K + khalf * 16;
  for (int ks = 0; ks < K; ks += 32) {
    v16bf a = load_a(arow, ks);
#pragma unroll
    for (int i = 0; i < 4; ++i) {
      const bf16_t* bp = Bw + (size_t)(n0base + i * 16 + l15) * K + ks + khalf * 16;
      acc[i] = wmma_bf16(a, *(const v16bf*)bp, acc[i]);
    }
  }

#pragma unroll
  for (int i = 0; i < 4; ++i) {
    int col = n0base + i * 16 + l15;
    float bv = bias[col];
#pragma unroll
    for (int r = 0; r < 8; ++r) {
      int row = m0 + khalf * 8 + r;
      float v = leaky_(acc[i][r] + bv);
      if (mode == 1) {
        int orow = (row & 31) * 20 + (row >> 5);           // (t*32+b) -> b*20+t
        out[(size_t)orow * N + col] = f2bf(v);
      } else {
        int c = col >> 4, y = (col >> 2) & 3, x = col & 3;  // (256,4,4) -> NHWC
        out[(((size_t)row * 4 + y) * 4 + x) * 256 + c] = f2bf(v);
      }
    }
  }
}

// ---------------------------------------------------------------------------
// Implicit-GEMM ConvTranspose2d (k=4,s=2,p=1) per output-parity class.
// A: NHWC bf16 activations (gathered, OOB taps -> 0). B: pre-packed bf16
// [parity][co][tap*Cin+ci]. Writes raw f32 NHWC + per-channel sum/sumsq.
// Block 256 = 8 waves x 16 rows (M tile 128), all N tiles per wave.
// ---------------------------------------------------------------------------
template <int LGC, int NT, int LGW>
__global__ void __launch_bounds__(256) k_deconv(
    const bf16_t* __restrict__ act, const bf16_t* __restrict__ bp,
    float* __restrict__ raw, float* __restrict__ stats)
{
  constexpr int Cin = 1 << LGC;
  constexpr int Cout = NT * 16;
  constexpr int W = 1 << LGW;      // Hin == Win == W
  constexpr int HW = W * W;
  constexpr int K4 = Cin * 4;

  const int p = blockIdx.y, py = p >> 1, px = p & 1;
  const int lane = threadIdx.x & 31, wave = threadIdx.x >> 5;
  const int khalf = lane >> 4, l15 = lane & 15;
  const int m0 = blockIdx.x * 128 + wave * 16;

  const int mA = m0 + l15;
  const int nA = mA >> (2 * LGW);
  const int remA = mA & (HW - 1);
  const int syA = remA >> LGW, sxA = remA & (W - 1);

  v8f zf = {};
  v8f acc[NT];
#pragma unroll
  for (int i = 0; i < NT; ++i) acc[i] = zf;

#pragma unroll
  for (int tap = 0; tap < 4; ++tap) {
    const int ty = tap >> 1, tx = tap & 1;
    const int dy = py ? (ty ? 0 : 1) : (ty ? -1 : 0);
    const int dx = px ? (tx ? 0 : 1) : (tx ? -1 : 0);
    const int iy = syA + dy, ix = sxA + dx;
    const bool valid = ((unsigned)iy < (unsigned)W) && ((unsigned)ix < (unsigned)W);
    const bf16_t* abase = act + ((size_t)nA * HW + (iy << LGW) + ix) * Cin + khalf * 16;
    for (int c0 = 0; c0 < Cin; c0 += 32) {
      v16bf a = zero16();
      if (valid) a = cat8(*(const v8bf*)(abase + c0), *(const v8bf*)(abase + c0 + 8));
      const int ks = tap * Cin + c0;
#pragma unroll
      for (int i = 0; i < NT; ++i) {
        const bf16_t* bptr = bp + ((size_t)p * Cout + (i * 16 + l15)) * K4 + ks + khalf * 16;
        acc[i] = wmma_bf16(a, *(const v16bf*)bptr, acc[i]);
      }
    }
  }

#pragma unroll
  for (int i = 0; i < NT; ++i) {
    const int co = i * 16 + l15;
    float s1 = 0.0f, s2 = 0.0f;
#pragma unroll
    for (int r = 0; r < 8; ++r) {
      const int m = m0 + khalf * 8 + r;
      const int n = m >> (2 * LGW);
      const int rem = m & (HW - 1);
      const int oy = ((rem >> LGW) << 1) + py;
      const int ox = ((rem & (W - 1)) << 1) + px;
      const float v = acc[i][r];
      raw[(((size_t)n * (2 * W) + oy) * (2 * W) + ox) * Cout + co] = v;
      s1 += v; s2 += v * v;
    }
    atomicAdd(&stats[co], s1);          // global_atomic_add_f32
    atomicAdd(&stats[Cout + co], s2);
  }
}

// BatchNorm (training-mode batch stats) + leaky, f32 -> bf16 NHWC.
__global__ void k_bn(const float* __restrict__ raw, const float* __restrict__ stats,
                     const float* __restrict__ g, const float* __restrict__ be,
                     bf16_t* __restrict__ act, int Cout, int total, float invCount)
{
  int i = blockIdx.x * 256 + threadIdx.x;
  if (i >= total) return;
  int c = i & (Cout - 1);
  float mean = stats[c] * invCount;
  float var  = stats[Cout + c] * invCount - mean * mean;   // biased var (ddof=0)
  float sc   = g[c] * rsqrtf(var + 1e-5f);
  act[i] = f2bf(leaky_((raw[i] - mean) * sc + be[c]));
}

// Final 32->1 ConvTranspose + sigmoid (N=1: direct VALU dot, memory bound).
__global__ void k_final(const bf16_t* __restrict__ act, const float* __restrict__ fwp,
                        const float* __restrict__ fb, float* __restrict__ out)
{
  int i = blockIdx.x * 256 + threadIdx.x;
  if (i >= 640 * 64 * 64) return;
  int n = i >> 12, oy = (i >> 6) & 63, ox = i & 63;
  int py = oy & 1, px = ox & 1, sy = oy >> 1, sx = ox >> 1;
  int p = py * 2 + px;
  float acc = fb[0];
#pragma unroll
  for (int t = 0; t < 4; ++t) {
    int ty = t >> 1, tx = t & 1;
    int dy = py ? (ty ? 0 : 1) : (ty ? -1 : 0);
    int dx = px ? (tx ? 0 : 1) : (tx ? -1 : 0);
    int iy = sy + dy, ix = sx + dx;
    if ((unsigned)iy < 32u && (unsigned)ix < 32u) {
      const bf16_t* ap = act + (((size_t)n * 32 + iy) * 32 + ix) * 32;
      const float* wp = fwp + (p * 4 + t) * 32;
#pragma unroll
      for (int ci = 0; ci < 32; ++ci) acc += bf2f(ap[ci]) * wp[ci];
    }
  }
  out[i] = 1.0f / (1.0f + __expf(-acc));
}

// ---------------------------------------------------------------------------
extern "C" void kernel_launch(void* const* d_in, const int* in_sizes, int n_in,
                              void* d_out, int out_size, void* d_ws, size_t ws_size,
                              hipStream_t stream) {
  const float* z           = (const float*)d_in[0];
  const float* fc_hidden_w = (const float*)d_in[1];
  const float* fc_hidden_b = (const float*)d_in[2];
  const float* fc_cell_w   = (const float*)d_in[3];
  const float* fc_cell_b   = (const float*)d_in[4];
  const float* w_ih        = (const float*)d_in[5];
  const float* b_ih        = (const float*)d_in[6];
  const float* w_hh        = (const float*)d_in[7];
  const float* b_hh        = (const float*)d_in[8];
  const float* init_inp    = (const float*)d_in[9];
  const float* fc_out_w    = (const float*)d_in[10];
  const float* fc_out_b    = (const float*)d_in[11];
  const float* cnn_fc_w    = (const float*)d_in[12];
  const float* cnn_fc_b    = (const float*)d_in[13];
  const float* dw0 = (const float*)d_in[14];
  const float* g0  = (const float*)d_in[15];
  const float* be0 = (const float*)d_in[16];
  const float* dw1 = (const float*)d_in[17];
  const float* g1  = (const float*)d_in[18];
  const float* be1 = (const float*)d_in[19];
  const float* dw2 = (const float*)d_in[20];
  const float* g2  = (const float*)d_in[21];
  const float* be2 = (const float*)d_in[22];
  const float* fw  = (const float*)d_in[23];
  const float* fb  = (const float*)d_in[24];
  float* out = (float*)d_out;

  char* ws = (char*)d_ws;
  size_t off = 0;
  auto alloc = [&](size_t bytes) {
    char* pp = ws + off;
    off += (bytes + 255) & ~(size_t)255;
    return pp;
  };

  bf16_t* whc   = (bf16_t*)alloc((size_t)512 * 256 * 2);
  bf16_t* wg    = (bf16_t*)alloc((size_t)1024 * 512 * 2);
  bf16_t* wout  = (bf16_t*)alloc((size_t)512 * 256 * 2);
  bf16_t* wfc   = (bf16_t*)alloc((size_t)4096 * 512 * 2);
  bf16_t* bp0   = (bf16_t*)alloc((size_t)4 * 128 * 1024 * 2);
  bf16_t* bp1   = (bf16_t*)alloc((size_t)4 * 64 * 512 * 2);
  bf16_t* bp2   = (bf16_t*)alloc((size_t)4 * 32 * 256 * 2);
  float*  fwp   = (float*)alloc((size_t)512 * 4);
  bf16_t* outs  = (bf16_t*)alloc((size_t)640 * 256 * 2);
  bf16_t* feats = (bf16_t*)alloc((size_t)640 * 512 * 2);
  bf16_t* act0  = (bf16_t*)alloc((size_t)640 * 16 * 256 * 2);
  bf16_t* act1  = (bf16_t*)alloc((size_t)640 * 64 * 128 * 2);
  bf16_t* act2  = (bf16_t*)alloc((size_t)640 * 256 * 64 * 2);
  bf16_t* act3  = (bf16_t*)alloc((size_t)640 * 1024 * 32 * 2);
  float*  raw   = (float*)alloc((size_t)640 * 1024 * 32 * 4);  // shared by all 3 layers
  float*  stats = (float*)alloc((size_t)256 * 4);

  // ---- weight packing (bf16) ----
  k_cvt<<<dim3(256), 256, 0, stream>>>(fc_hidden_w, whc, 256 * 256);
  k_cvt<<<dim3(256), 256, 0, stream>>>(fc_cell_w, whc + 256 * 256, 256 * 256);
  k_pack_wg<<<dim3(2048), 256, 0, stream>>>(w_ih, w_hh, wg);
  k_cvt<<<dim3(512), 256, 0, stream>>>(fc_out_w, wout, 512 * 256);
  k_cvt<<<dim3(8192), 256, 0, stream>>>(cnn_fc_w, wfc, 4096 * 512);
  k_pack_deconv<<<dim3(2048), 256, 0, stream>>>(dw0, bp0, 256, 128);
  k_pack_deconv<<<dim3(512), 256, 0, stream>>>(dw1, bp1, 128, 64);
  k_pack_deconv<<<dim3(128), 256, 0, stream>>>(dw2, bp2, 64, 32);
  k_pack_final<<<dim3(2), 256, 0, stream>>>(fw, fwp);

  // ---- persistent LSTM (1 WG, 32 waves, 192KB LDS) ----
  (void)hipFuncSetAttribute((const void*)k_lstm,
                            hipFuncAttributeMaxDynamicSharedMemorySize, 196608);
  k_lstm<<<dim3(1), 1024, 196608, stream>>>(z, init_inp, whc, fc_hidden_b, fc_cell_b,
                                            wg, b_ih, b_hh, outs);

  // feats = leaky(outs @ fc_out_w^T + b), with (t,b)->(b,t) row remap
  k_gemm<<<dim3(10, 4), 256, 0, stream>>>(outs, wout, fc_out_b, feats, 512, 256, 1);
  // cnn_fc = leaky(feats @ cnn_fc_w^T + b), stored NHWC (640,4,4,256)
  k_gemm<<<dim3(10, 32), 256, 0, stream>>>(feats, wfc, cnn_fc_b, act0, 4096, 512, 2);

  // ---- deconv stack: implicit GEMM + fused stats, then BN+leaky ----
  k_zero<<<dim3(1), 256, 0, stream>>>(stats, 256);
  k_deconv<8, 8, 2><<<dim3(80, 4), 256, 0, stream>>>(act0, bp0, raw, stats);
  k_bn<<<dim3(20480), 256, 0, stream>>>(raw, stats, g0, be0, act1, 128,
                                        5242880, 1.0f / 40960.0f);

  k_zero<<<dim3(1), 256, 0, stream>>>(stats, 256);
  k_deconv<7, 4, 3><<<dim3(320, 4), 256, 0, stream>>>(act1, bp1, raw, stats);
  k_bn<<<dim3(40960), 256, 0, stream>>>(raw, stats, g1, be1, act2, 64,
                                        10485760, 1.0f / 163840.0f);

  k_zero<<<dim3(1), 256, 0, stream>>>(stats, 256);
  k_deconv<6, 2, 4><<<dim3(1280, 4), 256, 0, stream>>>(act2, bp2, raw, stats);
  k_bn<<<dim3(81920), 256, 0, stream>>>(raw, stats, g2, be2, act3, 32,
                                        20971520, 1.0f / 655360.0f);

  // ---- final 1-channel deconv + sigmoid -> d_out (B,T,1,64,64) ----
  k_final<<<dim3(10240), 256, 0, stream>>>(act3, fwp, fb, out);
}
